// FeatureExtractor_80161269612904
// MI455X (gfx1250) — compile-verified, matrix operand order checked by reading, not run on hardware
//
#include <hip/hip_runtime.h>
#include <hip/hip_bf16.h>
#include <cstdint>

// ---------------------------------------------------------------------------
// MI455X (gfx1250) implementation: bf16 WMMA GEMM chain with fp32 accum/stats.
// A tiles staged to LDS via gfx1250 async Global->LDS copies (ASYNCcnt).
// ---------------------------------------------------------------------------

typedef __bf16 bf16_t;
typedef bf16_t v16bf __attribute__((ext_vector_type(16)));
typedef float  v8f   __attribute__((ext_vector_type(8)));

#define EPS    1e-5f
#define Bsz    16384
#define Dd     512
#define NC2    2048
#define INV_B  (1.0f / 16384.0f)

union Frag  { v16bf v; uint4 u[2]; };
union Pack4 { uint2 u; bf16_t b[4]; };

__device__ __forceinline__ float sigmoidf_(float x) { return 1.0f / (1.0f + __expf(-x)); }

// async 16B global -> LDS copy (per lane), tracked by ASYNCcnt
__device__ __forceinline__ void async_g2l_b128(void* lds_ptr, const void* gptr) {
  unsigned ldsa = (unsigned)(uintptr_t)lds_ptr;                 // LDS byte offset
  unsigned long long ga = (unsigned long long)(uintptr_t)gptr;  // 64-bit global addr
  asm volatile("global_load_async_to_lds_b128 %0, %1, off"
               :: "v"(ldsa), "v"(ga)
               : "memory");
}
__device__ __forceinline__ void async_wait0() {
  asm volatile("s_wait_asynccnt 0x0" ::: "memory");
}

// --------------------------------- utils -----------------------------------

__global__ void k_zero(float* p, int n) {
  int i = blockIdx.x * 256 + threadIdx.x;
  if (i < n) p[i] = 0.f;
}

// column sums/sumsq of x (for BN1 batch stats) + bf16 cast of x
__global__ __launch_bounds__(512) void k_colstats(const float* __restrict__ x,
                                                  bf16_t* __restrict__ xb,
                                                  float* __restrict__ accX) {
  const int col = threadIdx.x;                    // 0..511
  const long row0 = (long)blockIdx.x * 64;
  float s = 0.f, q = 0.f;
  for (int r = 0; r < 64; ++r) {
    float v = x[(row0 + r) * Dd + col];
    s += v; q += v * v;
    xb[(row0 + r) * Dd + col] = (bf16_t)v;
  }
  atomicAdd(&accX[col * 2 + 0], s);
  atomicAdd(&accX[col * 2 + 1], q);
}

// Fold BN1 scale into w1 and emit in WMMA B-fragment-swizzled bf16 layout:
// element index = (((kc*32 + ntg)*32 + lane)*16 + e)
// K = kc*32 + 16*(e>>3) + 8*(lane>>4) + (e&7) ; N = ntg*16 + (lane&15)
__global__ __launch_bounds__(256) void k_prep_w1(const float* __restrict__ w1,
                                                 const float* __restrict__ g1,
                                                 const float* __restrict__ accX,
                                                 bf16_t* __restrict__ bsw) {
  int idx = blockIdx.x * 256 + threadIdx.x;       // < 1048576
  int e = idx & 15, lane = (idx >> 4) & 31, ntg = (idx >> 9) & 31;
  int kc = (idx >> 14) & 15, d = idx >> 18;
  int K = kc * 32 + 16 * (e >> 3) + 8 * (lane >> 4) + (e & 7);
  int N = ntg * 16 + (lane & 15);
  float m   = accX[K * 2] * INV_B;
  float var = accX[K * 2 + 1] * INV_B - m * m;
  float sc  = g1[d * Dd + K] * rsqrtf(var + EPS);
  bsw[idx] = (bf16_t)(sc * w1[((long)(d * Dd + N)) * Dd + K]);
}

// bias1[d][o] = sum_j (bn1_b - m*sc) * w1[d][o][j]   (fp32, exact fold of BN1 shift)
__global__ __launch_bounds__(256) void k_bias1(const float* __restrict__ w1,
                                               const float* __restrict__ g1,
                                               const float* __restrict__ b1,
                                               const float* __restrict__ accX,
                                               float* __restrict__ bias1) {
  int d = blockIdx.x >> 9, o = blockIdx.x & 511;
  int tid = threadIdx.x;
  float p = 0.f;
  for (int j = tid; j < Dd; j += 256) {
    float m   = accX[j * 2] * INV_B;
    float var = accX[j * 2 + 1] * INV_B - m * m;
    float sc  = g1[d * Dd + j] * rsqrtf(var + EPS);
    float tb  = b1[d * Dd + j] - m * sc;
    p += tb * w1[((long)(d * Dd + o)) * Dd + j];
  }
#pragma unroll
  for (int msk = 1; msk < 32; msk <<= 1) p += __shfl_xor(p, msk, 32);
  __shared__ float wsum[8];
  if ((tid & 31) == 0) wsum[tid >> 5] = p;
  __syncthreads();
  if (tid == 0) {
    float t = 0.f;
    for (int w = 0; w < 8; ++w) t += wsum[w];
    bias1[blockIdx.x] = t;
  }
}

// Generic weight -> WMMA B-fragment swizzle (w is [N][K=512] row-major fp32)
__global__ __launch_bounds__(256) void k_prep_w(const float* __restrict__ w,
                                                bf16_t* __restrict__ bsw,
                                                int ntot, int nelem) {
  int idx = blockIdx.x * 256 + threadIdx.x;
  if (idx >= nelem) return;
  int e = idx & 15, lane = (idx >> 4) & 31;
  int rest = idx >> 9;
  int ntg = rest % ntot, kc = rest / ntot;
  int K = kc * 32 + 16 * (e >> 3) + 8 * (lane >> 4) + (e & 7);
  int N = ntg * 16 + (lane & 15);
  bsw[idx] = (bf16_t)w[(long)N * Dd + K];
}

// --------------------------- main fused GEMM -------------------------------
// Block: 256 threads (8 waves), 64 rows x 512 cols, K=512 in 16 chunks of 32.
// Wave w owns N-tiles 4w..4w+3 across all 4 M-tiles: 16 v8f accumulators.
// A tile is XOR-swizzled in 64KB LDS via async Global->LDS b128 copies:
// conflict-free ds_load_b128 fragment reads, no VGPR staging.
// B fragments come pre-swizzled from L2: two coalesced b128 loads per lane.
// mode: 0 = bias+store bf16 ; 1 = + column sum/sumsq atomics (BatchNorm)
//       2 = bias + residual + per-row LayerNorm, store bf16

struct RedSM {
  float rsum[64][8];
  float rsq [64][8];
  float mean[64];
  float rstd[64];
};

__global__ __launch_bounds__(256) void k_gemm512(
    const bf16_t* __restrict__ A, long a_stride,
    const bf16_t* __restrict__ Bsw, long b_zstride,
    const float* __restrict__ bias, int bias_zstride,
    bf16_t* __restrict__ outp, long out_stride, long out_zoff,
    float* __restrict__ statacc, int stat_zstride,
    const bf16_t* __restrict__ resid, long resid_stride,
    const float* __restrict__ ln_g, const float* __restrict__ ln_b,
    int mode) {
  __shared__ union { uint4 a4[64 * 64]; RedSM red; } sm;

  const int tid  = threadIdx.x;
  const int wave = tid >> 5;
  const int lane = tid & 31;
  const int h    = lane >> 4;
  const int l15  = lane & 15;
  const int z    = blockIdx.z;
  const long row0 = (long)blockIdx.x * 64;

  {  // async A tile stage: 64 rows x 512 bf16, xor-swizzled 16B chunks
    const bf16_t* abase = A + row0 * a_stride;
#pragma unroll
    for (int i = 0; i < 16; ++i) {
      int c = tid + i * 256;
      int r = c >> 6, cc = c & 63;
      async_g2l_b128(&sm.a4[r * 64 + (cc ^ (r & 15))],
                     abase + (long)r * a_stride + cc * 8);
    }
    async_wait0();
  }
  __syncthreads();

  const v8f vzero = {0.f, 0.f, 0.f, 0.f, 0.f, 0.f, 0.f, 0.f};
  v8f acc[4][4];
#pragma unroll
  for (int mt = 0; mt < 4; ++mt)
#pragma unroll
    for (int nt = 0; nt < 4; ++nt) acc[mt][nt] = vzero;

  const bf16_t* bz = Bsw + (long)z * b_zstride;
  for (int kc = 0; kc < 16; ++kc) {
    Frag af[4];
#pragma unroll
    for (int mt = 0; mt < 4; ++mt) {
      int r = mt * 16 + l15, sw = r & 15;
      af[mt].u[0] = sm.a4[r * 64 + ((kc * 4 + h) ^ sw)];
      af[mt].u[1] = sm.a4[r * 64 + ((kc * 4 + 2 + h) ^ sw)];
    }
#pragma unroll
    for (int nt = 0; nt < 4; ++nt) {
      int ntg = wave * 4 + nt;
      const uint4* bp = (const uint4*)(bz + ((((long)kc * 32 + ntg) * 32 + lane) << 4));
      Frag bfr; bfr.u[0] = bp[0]; bfr.u[1] = bp[1];
#pragma unroll
      for (int mt = 0; mt < 4; ++mt)
        acc[mt][nt] = __builtin_amdgcn_wmma_f32_16x16x32_bf16(
            false, af[mt].v, false, bfr.v, (short)0, acc[mt][nt], false, false);
    }
  }

  if (mode == 2) {
    __syncthreads();  // done with A tile LDS; reuse for row reductions
    // fold bias + residual into accumulators
#pragma unroll
    for (int mt = 0; mt < 4; ++mt)
#pragma unroll
      for (int nt = 0; nt < 4; ++nt) {
        int col = (wave * 4 + nt) * 16 + l15;
        float bb = bias[col];
#pragma unroll
        for (int reg = 0; reg < 8; ++reg) {
          long rowg = row0 + mt * 16 + reg + 8 * h;
          acc[mt][nt][reg] += bb + (float)resid[rowg * resid_stride + col];
        }
      }
    // per-row partials over this wave's 64 columns -> LDS
#pragma unroll
    for (int mt = 0; mt < 4; ++mt)
#pragma unroll
      for (int reg = 0; reg < 8; ++reg) {
        float s = 0.f, q = 0.f;
#pragma unroll
        for (int nt = 0; nt < 4; ++nt) { float v = acc[mt][nt][reg]; s += v; q += v * v; }
#pragma unroll
        for (int msk = 1; msk <= 8; msk <<= 1) { s += __shfl_xor(s, msk, 32); q += __shfl_xor(q, msk, 32); }
        if (l15 == 0) {
          int row = mt * 16 + reg + 8 * h;
          sm.red.rsum[row][wave] = s;
          sm.red.rsq [row][wave] = q;
        }
      }
    __syncthreads();
    if (tid < 64) {
      float s = 0.f, q = 0.f;
      for (int w = 0; w < 8; ++w) { s += sm.red.rsum[tid][w]; q += sm.red.rsq[tid][w]; }
      float m = s * (1.f / 512.f);
      float var = q * (1.f / 512.f) - m * m;
      sm.red.mean[tid] = m;
      sm.red.rstd[tid] = rsqrtf(var + EPS);
    }
    __syncthreads();
#pragma unroll
    for (int mt = 0; mt < 4; ++mt)
#pragma unroll
      for (int nt = 0; nt < 4; ++nt) {
        int col = (wave * 4 + nt) * 16 + l15;
        float g = ln_g[col], b = ln_b[col];
#pragma unroll
        for (int reg = 0; reg < 8; ++reg) {
          int rowL = mt * 16 + reg + 8 * h;
          long rowg = row0 + rowL;
          float y = (acc[mt][nt][reg] - sm.red.mean[rowL]) * sm.red.rstd[rowL] * g + b;
          outp[rowg * out_stride + col] = (bf16_t)y;
        }
      }
  } else {  // mode 0 / 1
#pragma unroll
    for (int nt = 0; nt < 4; ++nt) {
      int col = (wave * 4 + nt) * 16 + l15;
      float bb = bias[z * bias_zstride + col];
      float cs = 0.f, cq = 0.f;
#pragma unroll
      for (int mt = 0; mt < 4; ++mt)
#pragma unroll
        for (int reg = 0; reg < 8; ++reg) {
          long rowg = row0 + mt * 16 + reg + 8 * h;
          float v = acc[mt][nt][reg] + bb;
          outp[rowg * out_stride + (long)z * out_zoff + col] = (bf16_t)v;
          cs += v; cq += v * v;
        }
      if (mode == 1) {
        cs += __shfl_xor(cs, 16, 32);
        cq += __shfl_xor(cq, 16, 32);
        if (lane < 16) {
          float* st = statacc + ((long)z * stat_zstride + col) * 2;
          atomicAdd(st, cs);
          atomicAdd(st + 1, cq);
        }
      }
    }
  }
}

// -------------------------- BatchNorm + sigmoid ----------------------------

__global__ __launch_bounds__(256) void k_bnsig_bf(const bf16_t* __restrict__ in,
                                                  bf16_t* __restrict__ out,
                                                  const float* __restrict__ g,
                                                  const float* __restrict__ b,
                                                  const float* __restrict__ acc,
                                                  int colmask, long n) {
  for (long i = (long)blockIdx.x * 256 + threadIdx.x; i < n; i += (long)gridDim.x * 256) {
    int col = (int)(i & colmask);
    float m = acc[col * 2] * INV_B;
    float var = acc[col * 2 + 1] * INV_B - m * m;
    float y = ((float)in[i] - m) * rsqrtf(var + EPS) * g[col] + b[col];
    out[i] = (bf16_t)sigmoidf_(y);
  }
}

__global__ __launch_bounds__(256) void k_bnsig_f32(const float* __restrict__ in,
                                                   float* __restrict__ out,
                                                   const float* __restrict__ g,
                                                   const float* __restrict__ b,
                                                   const float* __restrict__ acc,
                                                   int colmask, long n) {
  for (long i = (long)blockIdx.x * 256 + threadIdx.x; i < n; i += (long)gridDim.x * 256) {
    int col = (int)(i & colmask);
    float m = acc[col * 2] * INV_B;
    float var = acc[col * 2 + 1] * INV_B - m * m;
    float y = (in[i] - m) * rsqrtf(var + EPS) * g[col] + b[col];
    out[i] = sigmoidf_(y);
  }
}

// ------------------------------ attention core -----------------------------
// One wave per (sample, head): S=4 tokens, head_dim=128 split 4 dims/lane.
__global__ __launch_bounds__(256) void k_attn(const bf16_t* __restrict__ q,
                                              const bf16_t* __restrict__ k,
                                              const bf16_t* __restrict__ v,
                                              bf16_t* __restrict__ o) {
  int gw = (int)(((long)blockIdx.x * 256 + threadIdx.x) >> 5);
  int lane = threadIdx.x & 31;
  long sample = gw >> 2;
  int head = gw & 3;
  long base = sample * (4 * Dd) + head * 128 + lane * 4;

  float qv[4][4], kv[4][4], vv[4][4];
#pragma unroll
  for (int t = 0; t < 4; ++t) {
    Pack4 a, bpk, c;
    a.u   = *(const uint2*)(q + base + t * Dd);
    bpk.u = *(const uint2*)(k + base + t * Dd);
    c.u   = *(const uint2*)(v + base + t * Dd);
#pragma unroll
    for (int e = 0; e < 4; ++e) {
      qv[t][e] = (float)a.b[e]; kv[t][e] = (float)bpk.b[e]; vv[t][e] = (float)c.b[e];
    }
  }
  float s[4][4];
#pragma unroll
  for (int i = 0; i < 4; ++i)
#pragma unroll
    for (int j = 0; j < 4; ++j) {
      float d = 0.f;
#pragma unroll
      for (int e = 0; e < 4; ++e) d += qv[i][e] * kv[j][e];
      s[i][j] = d;
    }
#pragma unroll
  for (int msk = 1; msk < 32; msk <<= 1)
#pragma unroll
    for (int i = 0; i < 4; ++i)
#pragma unroll
      for (int j = 0; j < 4; ++j) s[i][j] += __shfl_xor(s[i][j], msk, 32);

  const float scale = 0.08838834764831845f;  // 1/sqrt(128)
  float p[4][4];
#pragma unroll
  for (int i = 0; i < 4; ++i) {
    float mx = -1e30f;
#pragma unroll
    for (int j = 0; j < 4; ++j) mx = fmaxf(mx, s[i][j] * scale);
    float sum = 0.f;
#pragma unroll
    for (int j = 0; j < 4; ++j) { float e = __expf(s[i][j] * scale - mx); p[i][j] = e; sum += e; }
    float inv = 1.f / sum;
#pragma unroll
    for (int j = 0; j < 4; ++j) p[i][j] *= inv;
  }
#pragma unroll
  for (int i = 0; i < 4; ++i) {
    Pack4 r;
#pragma unroll
    for (int e = 0; e < 4; ++e) {
      float a = 0.f;
#pragma unroll
      for (int j = 0; j < 4; ++j) a += p[i][j] * vv[j][e];
      r.b[e] = (bf16_t)a;
    }
    *(uint2*)(o + base + i * Dd) = r.u;
  }
}

// ----------------------- EnsembleLinear2 (N=32) + BN3 stats ----------------
__global__ __launch_bounds__(128) void k_gemm_n32(const bf16_t* __restrict__ A,
                                                  const bf16_t* __restrict__ Bsw,
                                                  float* __restrict__ out,
                                                  float* __restrict__ statacc) {
  __shared__ uint4 a4[64 * 64];
  const int tid = threadIdx.x, wave = tid >> 5, lane = tid & 31;
  const int h = lane >> 4, l15 = lane & 15;
  const int z = blockIdx.z;
  const long s0 = (long)blockIdx.x * 64;

#pragma unroll
  for (int i = 0; i < 32; ++i) {
    int c = tid + i * 128;
    int r = c >> 6, cc = c & 63;
    async_g2l_b128(&a4[r * 64 + (cc ^ (r & 15))],
                   A + ((s0 + r) * 4 + z) * Dd + cc * 8);
  }
  async_wait0();
  __syncthreads();

  const v8f vzero = {0.f, 0.f, 0.f, 0.f, 0.f, 0.f, 0.f, 0.f};
  v8f acc[2] = {vzero, vzero};
  for (int kc = 0; kc < 16; ++kc) {
    Frag af;
    int r = wave * 16 + l15, sw = r & 15;
    af.u[0] = a4[r * 64 + ((kc * 4 + h) ^ sw)];
    af.u[1] = a4[r * 64 + ((kc * 4 + 2 + h) ^ sw)];
#pragma unroll
    for (int nt = 0; nt < 2; ++nt) {
      int ntg = z * 2 + nt;
      const uint4* bp = (const uint4*)(Bsw + ((((long)kc * 8 + ntg) * 32 + lane) << 4));
      Frag bfr; bfr.u[0] = bp[0]; bfr.u[1] = bp[1];
      acc[nt] = __builtin_amdgcn_wmma_f32_16x16x32_bf16(
          false, af.v, false, bfr.v, (short)0, acc[nt], false, false);
    }
  }
#pragma unroll
  for (int nt = 0; nt < 2; ++nt) {
    int col = (z * 2 + nt) * 16 + l15;
    float cs = 0.f, cq = 0.f;
#pragma unroll
    for (int reg = 0; reg < 8; ++reg) {
      long srow = s0 + wave * 16 + reg + 8 * h;
      float v = acc[nt][reg];
      out[srow * 128 + col] = v;
      cs += v; cq += v * v;
    }
    cs += __shfl_xor(cs, 16, 32);
    cq += __shfl_xor(cq, 16, 32);
    if (lane < 16) {
      atomicAdd(&statacc[col * 2], cs);
      atomicAdd(&statacc[col * 2 + 1], cq);
    }
  }
}

// ------------------------------- host driver -------------------------------

extern "C" void kernel_launch(void* const* d_in, const int* in_sizes, int n_in,
                              void* d_out, int out_size, void* d_ws, size_t ws_size,
                              hipStream_t stream) {
  (void)in_sizes; (void)n_in; (void)out_size; (void)ws_size;
  const float* x    = (const float*)d_in[0];
  const float* bn1g = (const float*)d_in[1];
  const float* bn1b = (const float*)d_in[2];
  const float* w1   = (const float*)d_in[3];
  const float* bn2g = (const float*)d_in[4];
  const float* bn2b = (const float*)d_in[5];
  const float* wq   = (const float*)d_in[6];
  const float* bq   = (const float*)d_in[7];
  const float* wk   = (const float*)d_in[8];
  const float* bk   = (const float*)d_in[9];
  const float* wv   = (const float*)d_in[10];
  const float* bv   = (const float*)d_in[11];
  const float* wo   = (const float*)d_in[12];
  const float* bo   = (const float*)d_in[13];
  const float* ln1g = (const float*)d_in[14];
  const float* ln1b = (const float*)d_in[15];
  const float* wff  = (const float*)d_in[16];
  const float* bff  = (const float*)d_in[17];
  const float* ln2g = (const float*)d_in[18];
  const float* ln2b = (const float*)d_in[19];
  const float* w2   = (const float*)d_in[20];
  const float* bn3g = (const float*)d_in[21];
  const float* bn3b = (const float*)d_in[22];
  float* outp = (float*)d_out;

  char* ws = (char*)d_ws;
  size_t off = 0;
  auto take = [&](size_t bytes) -> char* {
    char* p = ws + off;
    off += (bytes + 255) & ~(size_t)255;
    return p;
  };

  float*  accX  = (float*)take(1024 * 4);                    // BN1 sums [512][2]
  float*  acc2  = (float*)take(4096 * 4);                    // BN2 sums [2048][2]
  float*  acc3  = (float*)take(256 * 4);                     // BN3 sums [128][2]
  bf16_t* xb    = (bf16_t*)take((size_t)Bsz * Dd * 2);
  float*  bias1 = (float*)take(2048 * 4);
  bf16_t* bsw1  = (bf16_t*)take((size_t)4 * 262144 * 2);
  bf16_t* bswq  = (bf16_t*)take(262144 * 2);
  bf16_t* bswk  = (bf16_t*)take(262144 * 2);
  bf16_t* bswv  = (bf16_t*)take(262144 * 2);
  bf16_t* bswo  = (bf16_t*)take(262144 * 2);
  bf16_t* bswf  = (bf16_t*)take(262144 * 2);
  bf16_t* bsw2  = (bf16_t*)take(65536 * 2);
  bf16_t* hb    = (bf16_t*)take((size_t)Bsz * NC2 * 2);      // h, then tb in-place
  bf16_t* qb    = (bf16_t*)take((size_t)Bsz * 4 * Dd * 2);
  bf16_t* kb    = (bf16_t*)take((size_t)Bsz * 4 * Dd * 2);
  bf16_t* vb    = (bf16_t*)take((size_t)Bsz * 4 * Dd * 2);
  bf16_t* ob    = (bf16_t*)take((size_t)Bsz * 4 * Dd * 2);
  bf16_t* a1 = qb;            // dead after attention
  bf16_t* a2 = kb;            // dead after attention
  float*  h2 = (float*)vb;    // 8MB into vb's 64MB, dead after attention

  // stats accumulators are contiguous: zero all 5376 floats
  k_zero<<<21, 256, 0, stream>>>(accX, 5376);

  // BN1 stats + bf16 cast of x
  k_colstats<<<256, 512, 0, stream>>>(x, xb, accX);

  // fold BN1 into w1 (swizzled bf16) + fp32 bias; swizzle remaining weights
  k_prep_w1<<<4096, 256, 0, stream>>>(w1, bn1g, accX, bsw1);
  k_bias1<<<2048, 256, 0, stream>>>(w1, bn1g, bn1b, accX, bias1);
  k_prep_w<<<1024, 256, 0, stream>>>(wq,  bswq, 32, 262144);
  k_prep_w<<<1024, 256, 0, stream>>>(wk,  bswk, 32, 262144);
  k_prep_w<<<1024, 256, 0, stream>>>(wv,  bswv, 32, 262144);
  k_prep_w<<<1024, 256, 0, stream>>>(wo,  bswo, 32, 262144);
  k_prep_w<<<1024, 256, 0, stream>>>(wff, bswf, 32, 262144);
  k_prep_w<<<256,  256, 0, stream>>>(w2,  bsw2, 8, 65536);

  // stage 1: h = (BN1-folded) EnsembleLinear1, accumulate BN2 batch stats
  k_gemm512<<<dim3(256, 1, 4), 256, 0, stream>>>(
      xb, 512, bsw1, 262144, bias1, 512, hb, 2048, 512, acc2, 512,
      nullptr, 0, nullptr, nullptr, 1);

  // BN2 + sigmoid (in place: h -> t tokens)
  k_bnsig_bf<<<16384, 256, 0, stream>>>(hb, hb, bn2g, bn2b, acc2, NC2 - 1,
                                        (long)Bsz * NC2);

  // q, k, v projections over [65536, 512]
  k_gemm512<<<dim3(1024, 1, 1), 256, 0, stream>>>(
      hb, 512, bswq, 0, bq, 0, qb, 512, 0, nullptr, 0, nullptr, 0, nullptr, nullptr, 0);
  k_gemm512<<<dim3(1024, 1, 1), 256, 0, stream>>>(
      hb, 512, bswk, 0, bk, 0, kb, 512, 0, nullptr, 0, nullptr, 0, nullptr, nullptr, 0);
  k_gemm512<<<dim3(1024, 1, 1), 256, 0, stream>>>(
      hb, 512, bswv, 0, bv, 0, vb, 512, 0, nullptr, 0, nullptr, 0, nullptr, nullptr, 0);

  // 4-token attention, one wave per (sample, head)
  k_attn<<<8192, 256, 0, stream>>>(qb, kb, vb, ob);

  // wo + bo + residual(t) -> LayerNorm1 -> a1
  k_gemm512<<<dim3(1024, 1, 1), 256, 0, stream>>>(
      ob, 512, bswo, 0, bo, 0, a1, 512, 0, nullptr, 0, hb, 512, ln1g, ln1b, 2);

  // wff + bff + residual(a1) -> LayerNorm2 -> a2
  k_gemm512<<<dim3(1024, 1, 1), 256, 0, stream>>>(
      a1, 512, bswf, 0, bff, 0, a2, 512, 0, nullptr, 0, a1, 512, ln2g, ln2b, 2);

  // EnsembleLinear2 (per-detector D->32) + BN3 stats
  k_gemm_n32<<<dim3(256, 1, 4), 128, 0, stream>>>(a2, bsw2, h2, acc3);

  // BN3 + sigmoid -> fp32 output
  k_bnsig_f32<<<4096, 256, 0, stream>>>(h2, outp, bn3g, bn3b, acc3, 127,
                                        (long)Bsz * 128);
}
